// DeformableConvLayer4_71047349010936
// MI455X (gfx1250) — matrix operand order repaired and checked by reading, not compiled, not run on hardware
//
#include <hip/hip_runtime.h>
#include <hip/hip_bf16.h>

// ---------------------------------------------------------------------------
// DeformableConvLayer on MI455X (gfx1250, wave32, WMMA).
// All four convolutions are executed as f16 WMMA GEMMs (f32 accumulate):
//   regular 3x3 conv == deformable conv with offset=0, mask=1.
// A (gathered im2col tile) is double-buffered in LDS per 32-wide K slice;
// B (weights) are pre-swizzled into the exact 16x16x32 B-fragment VGPR
// layout so each lane fetches its fragment with two contiguous b128 loads.
// Wave-level tile guards use readfirstlane so WMMA always runs with
// EXEC == all-ones (scalar branch, no saveexec).
// ---------------------------------------------------------------------------

typedef __attribute__((ext_vector_type(16))) _Float16     v16h;
typedef __attribute__((ext_vector_type(8)))  float        v8f;
typedef __attribute__((ext_vector_type(4)))  float        v4f;
typedef __attribute__((ext_vector_type(4)))  unsigned int v4u;

union Frag16 { v16h h; v4u u[2]; };
union PackH  { _Float16 h[2]; unsigned int u; };

constexpr int Hc  = 128;
constexpr int Wc  = 128;
constexpr int HWc = Hc * Wc;      // 16384
constexpr int Bc  = 2;

// ---------------------------------------------------------------------------
// Weight swizzle: (O, C*9) f32 row-major (flat index r = c*9 + k, matching
// weight.reshape(O, C, 3, 3)) -> f16 packed in B-fragment order:
//   dword index ((s*ntiles + t)*32 + L)*8 + v holds
//   K = s*32 + (v>=4?16:0) + (L>=16?8:0) + 2*(v&3) + {0,1},  o = t*16 + (L&15).
// Supports concatenating two weight tensors along O (offset conv + mod conv)
// and zero-pads o >= Oreal (padded N columns must be deterministic zeros).
// ---------------------------------------------------------------------------
__global__ void swz_kernel(const float* __restrict__ w1, int O1,
                           const float* __restrict__ w2, int O2,
                           int C, int ntiles, unsigned int* __restrict__ out)
{
    const int Ktot = C * 9;
    const int idx  = blockIdx.x * 256 + threadIdx.x;   // grid sized exactly
    const int v  = idx & 7;
    const int L  = (idx >> 3) & 31;
    const int st = idx >> 8;
    const int t  = st % ntiles;
    const int s  = st / ntiles;
    const int o  = t * 16 + (L & 15);
    const int K0 = s * 32 + ((v >> 2) << 4) + ((L >> 4) << 3) + 2 * (v & 3);

    float f0 = 0.f, f1 = 0.f;
    const int Oreal = O1 + O2;
    if (o < Oreal) {
        const float* wr = (o < O1) ? (w1 + (size_t)o * Ktot)
                                   : (w2 + (size_t)(o - O1) * Ktot);
        f0 = wr[K0];
        f1 = wr[K0 + 1];
    }
    PackH p;
    p.h[0] = (_Float16)f0;
    p.h[1] = (_Float16)f1;
    out[idx] = p.u;
}

// ---------------------------------------------------------------------------
// Fused deformable-im2col + WMMA GEMM.
//   x     : (B, C, H, W) f32 input
//   wswz  : swizzled f16 weights, ntiles = Opad/16, Ksteps = C*9/32
//   om    : (B, 27, HW) offset/mask conv output, or nullptr => offsets=0,
//           mask=1 (plain 3x3 conv, pad 1). Channels: 2k=dy, 2k+1=dx,
//           18+k = raw modulation (2*sigmoid applied here at read time).
//   bias1/bias2/bsplit: epilogue bias = o<bsplit ? bias1[o] : bias2[o-bsplit]
//   out   : (B, Oreal, HW) f32
// One workgroup (128 thr = 4 waves) owns 16 consecutive spatial positions;
// wave w computes N-tiles {w, w+4, ...} (NT of them, compile-time).
// ---------------------------------------------------------------------------
template <int NT>
__global__ __launch_bounds__(128)
void deform_gemm_kernel(const float* __restrict__ x, int C,
                        const unsigned int* __restrict__ wswz,
                        int ntiles, int Oreal,
                        const float* __restrict__ om,
                        const float* __restrict__ bias1,
                        const float* __restrict__ bias2, int bsplit,
                        float* __restrict__ out)
{
    __shared__ float s_py[144];
    __shared__ float s_px[144];
    __shared__ float s_mv[144];
    __shared__ __align__(16) _Float16 s_a[2][16 * 40]; // double-buffered A tile

    const int tid  = threadIdx.x;
    const int pos0 = blockIdx.x * 16;          // 16 spatial positions / block
    const int b    = pos0 / HWc;               // tiles never straddle batch
    const int hw0  = pos0 % HWc;
    const int Ksteps = (C * 9) / 32;           // 576/32=18, 2304/32=72 exact

    // ---- stage per-(position, tap) sampling coords + modulation ----
    for (int i = tid; i < 144; i += 128) {
        const int m  = i / 9;
        const int k  = i - m * 9;
        const int hw = hw0 + m;
        const int ho = hw >> 7;                // W = 128
        const int wo = hw & 127;
        float dy = 0.f, dx = 0.f, mv = 1.f;
        if (om) {
            const float* omb = om + (size_t)b * 27 * HWc + hw;
            dy = omb[(size_t)(2 * k) * HWc];
            dx = omb[(size_t)(2 * k + 1) * HWc];
            const float mm = omb[(size_t)(18 + k) * HWc];
            mv = 2.f / (1.f + __expf(-mm));    // 2 * sigmoid
        }
        s_py[i] = dy + (float)(k / 3) + (float)(ho - 1);
        s_px[i] = dx + (float)(k % 3) + (float)(wo - 1);
        s_mv[i] = mv;
    }

    const float* xb = x + (size_t)b * C * HWc;

    // Gather one 16x32 f16 A slice (K rows r = ks*32 .. +31) into s_a[buf].
    auto stageA = [&](int ks, int buf) {
#pragma unroll
        for (int i = 0; i < 4; ++i) {
            const int e  = tid + i * 128;      // 512 elements / 128 threads
            const int m  = e >> 5;
            const int kl = e & 31;
            const int r  = ks * 32 + kl;       // r = c*9 + k
            const int c  = r / 9;
            const int k  = r - c * 9;
            const int ck = m * 9 + k;
            const float py = s_py[ck], px = s_px[ck], mv = s_mv[ck];
            const float fy = floorf(py), fx = floorf(px);
            const int   y0 = (int)fy,    x0 = (int)fx;
            const float ly = py - fy,    lx = px - fx;
            const float* xc = xb + (size_t)c * HWc;
            const bool yi0 = (y0 >= 0) && (y0 < Hc);
            const bool yi1 = (y0 + 1 >= 0) && (y0 + 1 < Hc);
            const bool xi0 = (x0 >= 0) && (x0 < Wc);
            const bool xi1 = (x0 + 1 >= 0) && (x0 + 1 < Wc);
            float v00 = 0.f, v01 = 0.f, v10 = 0.f, v11 = 0.f;
            if (yi0 && xi0) v00 = xc[y0 * Wc + x0];
            if (yi0 && xi1) v01 = xc[y0 * Wc + x0 + 1];
            if (yi1 && xi0) v10 = xc[(y0 + 1) * Wc + x0];
            if (yi1 && xi1) v11 = xc[(y0 + 1) * Wc + x0 + 1];
            const float val = v00 * (1.f - ly) * (1.f - lx)
                            + v01 * (1.f - ly) * lx
                            + v10 * ly * (1.f - lx)
                            + v11 * ly * lx;
            s_a[buf][m * 40 + kl] = (_Float16)(val * mv);
        }
    };

    v8f acc[NT];
#pragma unroll
    for (int t = 0; t < NT; ++t)
#pragma unroll
        for (int e = 0; e < 8; ++e) acc[t][e] = 0.f;

    // Scalarize the wave id so tile guards are s_cbranch (EXEC untouched
    // around WMMA, which requires EXEC == all-ones).
    const int wave   = __builtin_amdgcn_readfirstlane(tid >> 5);
    const int lane   = tid & 31;
    const int lanehi = lane >> 4;

    __syncthreads();          // coords visible
    stageA(0, 0);             // prologue: first A slice
    __syncthreads();

    for (int ks = 0; ks < Ksteps; ++ks) {
        const int cur = ks & 1;

        // ---- A fragment: lane holds M=lane%16;
        //      K(v,h) = (v>=4?16:0) + (lane>=16?8:0) + (v%4)*2 + h
        //      => two contiguous 16B LDS reads (ds_load_b128 x2).
        Frag16 a;
        {
            const _Float16* arow = &s_a[cur][(lane & 15) * 40 + lanehi * 8];
            a.u[0] = *(const v4u*)(arow);        // K = lanehi*8 .. +7
            a.u[1] = *(const v4u*)(arow + 16);   // K = 16+lanehi*8 .. +7
        }

        // Pull next K-slice of B toward the WGP (global_prefetch_b8).
        if (ks + 1 < Ksteps && wave < ntiles) {
            __builtin_prefetch((const void*)(wswz +
                (((size_t)(ks + 1) * ntiles + wave) * 32 + lane) * 8), 0, 0);
        }

        // ---- B fragments (pre-swizzled) + WMMA ----
#pragma unroll
        for (int tt = 0; tt < NT; ++tt) {
            const int tile = wave + 4 * tt;      // scalar guard
            if (tile < ntiles) {
                const v4u* bp = (const v4u*)(wswz +
                    (((size_t)ks * ntiles + tile) * 32 + lane) * 8);
                Frag16 bf;
                bf.u[0] = bp[0];
                bf.u[1] = bp[1];
                acc[tt] = __builtin_amdgcn_wmma_f32_16x16x32_f16(
                    false, a.h, false, bf.h, (short)0, acc[tt], false, false);
            }
        }

        // ---- overlap: gather next slice into the other buffer ----
        if (ks + 1 < Ksteps) stageA(ks + 1, cur ^ 1);
        __syncthreads();   // readers of cur done + next slice staged
    }

    // ---- epilogue: D layout lane->(N=lane%16, M=(lane>=16?8:0)+r) ----
#pragma unroll
    for (int tt = 0; tt < NT; ++tt) {
        const int tile = wave + 4 * tt;
        if (tile >= ntiles) continue;            // scalar branch
        const int o = tile * 16 + (lane & 15);
        if (o >= Oreal) continue;                // vector guard (post-WMMA, ok)
        float bias = 0.f;
        if (bias1) bias = (o < bsplit) ? bias1[o] : bias2[o - bsplit];
        float* op = out + ((size_t)b * Oreal + o) * HWc + hw0 + lanehi * 8;
        v4f lo = { acc[tt][0] + bias, acc[tt][1] + bias,
                   acc[tt][2] + bias, acc[tt][3] + bias };
        v4f hi = { acc[tt][4] + bias, acc[tt][5] + bias,
                   acc[tt][6] + bias, acc[tt][7] + bias };
        *(v4f*)op       = lo;   // 8 consecutive spatial positions -> 2x b128
        *(v4f*)(op + 4) = hi;
    }
}

// ---------------------------------------------------------------------------
// In-place channel LayerNorm + exact GELU. One thread per (b, h, w) pixel.
// Whole tensors are L2-resident (192 MB) so the strided channel walk is fine.
// ---------------------------------------------------------------------------
__global__ void ln_gelu_kernel(float* __restrict__ p,
                               const float* __restrict__ w,
                               const float* __restrict__ bta, int Cch)
{
    const int pix = blockIdx.x * blockDim.x + threadIdx.x;  // B*HW threads
    const int bN  = pix / HWc;
    const int hw  = pix % HWc;
    float* base = p + (size_t)bN * Cch * HWc + hw;

    float sum = 0.f;
    for (int c = 0; c < Cch; ++c) sum += base[(size_t)c * HWc];
    const float mean = sum / (float)Cch;

    float s2 = 0.f;
    for (int c = 0; c < Cch; ++c) {
        const float d = base[(size_t)c * HWc] - mean;
        s2 += d * d;
    }
    const float inv = rsqrtf(s2 / (float)Cch + 1e-5f);

    for (int c = 0; c < Cch; ++c) {
        const float xh = (base[(size_t)c * HWc] - mean) * inv;
        const float y  = w[c] * xh + bta[c];
        base[(size_t)c * HWc] = 0.5f * y * (1.f + erff(y * 0.70710678118654752f));
    }
}

// ---------------------------------------------------------------------------
extern "C" void kernel_launch(void* const* d_in, const int* in_sizes, int n_in,
                              void* d_out, int out_size, void* d_ws, size_t ws_size,
                              hipStream_t stream)
{
    const float* x      = (const float*)d_in[0];
    const float* off1_w = (const float*)d_in[1];
    const float* off1_b = (const float*)d_in[2];
    const float* mod1_w = (const float*)d_in[3];
    const float* mod1_b = (const float*)d_in[4];
    const float* reg1_w = (const float*)d_in[5];
    const float* ln1_w  = (const float*)d_in[6];
    const float* ln1_b  = (const float*)d_in[7];
    const float* off2_w = (const float*)d_in[8];
    const float* off2_b = (const float*)d_in[9];
    const float* mod2_w = (const float*)d_in[10];
    const float* mod2_b = (const float*)d_in[11];
    const float* reg2_w = (const float*)d_in[12];
    const float* ln2_w  = (const float*)d_in[13];
    const float* ln2_b  = (const float*)d_in[14];
    float* out = (float*)d_out;

    // ---- workspace carve-up (floats) ----
    float* ws  = (float*)d_ws;
    float* om1 = ws;                       // (2, 27, 16384) = 884736
    float* om2 = om1 + 884736;             // 884736
    float* h1  = om2 + 884736;             // (2, 256, 16384) = 8388608
    unsigned int* wc1 = (unsigned int*)(h1 + 8388608);  // 18*2*256  =  9216 dw
    unsigned int* wr1 = wc1 + 9216;                     // 18*16*256 = 73728 dw
    unsigned int* wc2 = wr1 + 73728;                    // 72*2*256  = 36864 dw
    unsigned int* wr2 = wc2 + 36864;                    // 72*4*256  = 73728 dw

    // ---- swizzle weights into B-fragment layout (f16) ----
    swz_kernel<<<9216 / 256, 256, 0, stream>>>(off1_w, 18, mod1_w, 9, 64, 2, wc1);
    swz_kernel<<<73728 / 256, 256, 0, stream>>>(reg1_w, 256, nullptr, 0, 64, 16, wr1);
    swz_kernel<<<36864 / 256, 256, 0, stream>>>(off2_w, 18, mod2_w, 9, 256, 2, wc2);
    swz_kernel<<<73728 / 256, 256, 0, stream>>>(reg2_w, 64, nullptr, 0, 256, 4, wr2);

    const int nblk = (Bc * HWc) / 16;   // 2048 workgroups of 128 threads

    // ---- block 1 ----
    deform_gemm_kernel<1><<<nblk, 128, 0, stream>>>(
        x, 64, wc1, 2, 27, nullptr, off1_b, mod1_b, 18, om1);
    deform_gemm_kernel<4><<<nblk, 128, 0, stream>>>(
        x, 64, wr1, 16, 256, om1, nullptr, nullptr, 0, h1);
    ln_gelu_kernel<<<(Bc * HWc) / 256, 256, 0, stream>>>(h1, ln1_w, ln1_b, 256);

    // ---- block 2 ----
    deform_gemm_kernel<1><<<nblk, 128, 0, stream>>>(
        h1, 256, wc2, 2, 27, nullptr, off2_b, mod2_b, 18, om2);
    deform_gemm_kernel<1><<<nblk, 128, 0, stream>>>(
        h1, 256, wr2, 4, 64, om2, nullptr, nullptr, 0, out);
    ln_gelu_kernel<<<(Bc * HWc) / 256, 256, 0, stream>>>(out, ln2_w, ln2_b, 64);
}
